// Pool_56633438765196
// MI455X (gfx1250) — 8-bit matrix operand order repaired and checked by reading, compiled, not run on hardware
//
#include <hip/hip_runtime.h>
#include <hip/hip_bf16.h>

#define NNODES 8192
#define INDIM  256
#define NATT   4
#define KKEEP  4096

typedef int v8i_t __attribute__((ext_vector_type(8)));

__global__ __launch_bounds__(256) void k_binarize(const float* __restrict__ g,
                                                  unsigned char* __restrict__ A) {
    const long long t = (long long)blockIdx.x * 256 + threadIdx.x;
    const int i = (int)(t >> 13);
    const int j = (int)(t & (NNODES - 1));
    const float v = g[t];
    A[t] = (unsigned char)((v != 0.0f) || (i == j));
}

__global__ __launch_bounds__(256) void k_transpose_u8(const unsigned char* __restrict__ src,
                                                      unsigned char* __restrict__ dst) {
    __shared__ unsigned char tile[32][33];
    const int bx = blockIdx.x, by = blockIdx.y;
    int x = bx * 32 + threadIdx.x;
    int y = by * 32 + threadIdx.y;
#pragma unroll
    for (int i = 0; i < 32; i += 8)
        tile[threadIdx.y + i][threadIdx.x] = src[(size_t)(y + i) * NNODES + x];
    __syncthreads();
    x = by * 32 + threadIdx.x;
    y = bx * 32 + threadIdx.y;
#pragma unroll
    for (int i = 0; i < 32; i += 8)
        dst[(size_t)(y + i) * NNODES + x] = tile[threadIdx.x][threadIdx.y + i];
}

__device__ __forceinline__ v8i_t load_a_frag(const unsigned char* pA, int k, int half) {
    const unsigned long long a0 = *(const unsigned long long*)(pA + k +  0 + 8 * half);
    const unsigned long long a1 = *(const unsigned long long*)(pA + k + 16 + 8 * half);
    const unsigned long long a2 = *(const unsigned long long*)(pA + k + 32 + 8 * half);
    const unsigned long long a3 = *(const unsigned long long*)(pA + k + 48 + 8 * half);
    v8i_t v;
    v[0] = (int)a0; v[1] = (int)(a0 >> 32);
    v[2] = (int)a1; v[3] = (int)(a1 >> 32);
    v[4] = (int)a2; v[5] = (int)(a2 >> 32);
    v[6] = (int)a3; v[7] = (int)(a3 >> 32);
    return v;
}

__device__ __forceinline__ v8i_t load_b_frag(const unsigned char* pB, int k, int half) {
    const unsigned long long b0 = *(const unsigned long long*)(pB + k +  0 + 8 * half);
    const unsigned long long b1 = *(const unsigned long long*)(pB + k + 16 + 8 * half);
    const unsigned long long b2 = *(const unsigned long long*)(pB + k + 32 + 8 * half);
    const unsigned long long b3 = *(const unsigned long long*)(pB + k + 48 + 8 * half);
    v8i_t v = { (int)b0, (int)(b0 >> 32), (int)b1, (int)(b1 >> 32), (int)b2, (int)(b2 >> 32), (int)b3, (int)(b3 >> 32) };
    return v;
}

__device__ __forceinline__ void store_c_bool(unsigned char* pC, const v8i_t& acc) {
#pragma unroll
    for (int r = 0; r < 8; ++r)
        pC[(size_t)r * NNODES] = (unsigned char)(acc[r] != 0);
}

__global__ __launch_bounds__(256) void k_bmm_bool(const unsigned char* __restrict__ A,
                                                  const unsigned char* __restrict__ BT,
                                                  unsigned char* __restrict__ Cout) {
    const int lane = threadIdx.x & 31;
    const int wave = threadIdx.x >> 5;
    const int tilesPerRow = NNODES / 32;
    const int flat = blockIdx.x * 8 + wave;
    const int rm = (flat / tilesPerRow) * 32;
    const int cn = (flat % tilesPerRow) * 32;

    const int m    = lane & 15;
    const int half = lane >> 4;

    const unsigned char* pA0 = A  + (size_t)(rm + m) * NNODES;
    const unsigned char* pA1 = pA0 + (size_t)16 * NNODES;
    const unsigned char* pB0 = BT + (size_t)(cn + m) * NNODES;
    const unsigned char* pB1 = pB0 + (size_t)16 * NNODES;

    v8i_t acc00 = {}, acc01 = {}, acc10 = {}, acc11 = {};

#pragma unroll 2
    for (int k = 0; k < NNODES; k += 64) {
        __builtin_prefetch(pA0 + k + 256, 0, 3);
        __builtin_prefetch(pA1 + k + 256, 0, 3);
        __builtin_prefetch(pB0 + k + 256, 0, 3);
        __builtin_prefetch(pB1 + k + 256, 0, 3);

        const v8i_t va0 = load_a_frag(pA0, k, half);
        const v8i_t va1 = load_a_frag(pA1, k, half);
        const v8i_t vb0 = load_b_frag(pB0, k, half);
        const v8i_t vb1 = load_b_frag(pB1, k, half);

        acc00 = __builtin_amdgcn_wmma_i32_16x16x64_iu8(false, va0, false, vb0, acc00, false, false);
        acc01 = __builtin_amdgcn_wmma_i32_16x16x64_iu8(false, va0, false, vb1, acc01, false, false);
        acc10 = __builtin_amdgcn_wmma_i32_16x16x64_iu8(false, va1, false, vb0, acc10, false, false);
        acc11 = __builtin_amdgcn_wmma_i32_16x16x64_iu8(false, va1, false, vb1, acc11, false, false);
    }

    unsigned char* pC = Cout + (size_t)(rm + 8 * half) * NNODES + cn + m;
    store_c_bool(pC, acc00);
    store_c_bool(pC + 16, acc01);
    store_c_bool(pC + (size_t)16 * NNODES, acc10);
    store_c_bool(pC + (size_t)16 * NNODES + 16, acc11);
}

__global__ __launch_bounds__(256) void k_score(const float* __restrict__ h,
                                               const float* __restrict__ W,
                                               const float* __restrict__ b,
                                               float* __restrict__ score) {
    const int lane = threadIdx.x & 31;
    const int wave = threadIdx.x >> 5;
    const int row  = blockIdx.x * 8 + wave;
    const float* hr = h + (size_t)row * INDIM;

    float acc[NATT] = {0.f, 0.f, 0.f, 0.f};
    for (int t = lane; t < INDIM; t += 32) {
        const float hv = hr[t];
#pragma unroll
        for (int a = 0; a < NATT; ++a) acc[a] += hv * W[a * INDIM + t];
    }
#pragma unroll
    for (int off = 16; off > 0; off >>= 1)
#pragma unroll
        for (int a = 0; a < NATT; ++a) acc[a] += __shfl_xor(acc[a], off, 32);

    if (lane == 0) {
        float s = 0.f;
#pragma unroll
        for (int a = 0; a < NATT; ++a)
            s += 1.0f / (1.0f + __expf(-(acc[a] + b[a])));
        score[row] = s;
    }
}

__global__ __launch_bounds__(256) void k_topk(const float* __restrict__ score,
                                              int* __restrict__ idxb,
                                              float* __restrict__ valb,
                                              float* __restrict__ out_idx) {
    __shared__ float sh[256];
    const int i = blockIdx.x * 256 + threadIdx.x;
    const float si = score[i];
    int r = 0;
    for (int t = 0; t < NNODES; t += 256) {
        sh[threadIdx.x] = score[t + threadIdx.x];
        __syncthreads();
#pragma unroll 8
        for (int j = 0; j < 256; ++j) {
            const float sj = sh[j];
            const int jj = t + j;
            r += (sj > si) || ((sj == si) && (jj < i));
        }
        __syncthreads();
    }
    if (r < KKEEP) {
        idxb[r] = i;
        valb[r] = si;
        out_idx[r] = (float)i;
    }
}

__global__ __launch_bounds__(256) void k_newh(const float* __restrict__ h,
                                              const int* __restrict__ idxb,
                                              const float* __restrict__ valb,
                                              float* __restrict__ out_h) {
    const int t = blockIdx.x * 256 + threadIdx.x;
    const int r = t >> 8;
    const int c = t & 255;
    out_h[t] = h[(size_t)idxb[r] * INDIM + c] * valb[r];
}

__global__ __launch_bounds__(256) void k_norm_gather(const unsigned char* __restrict__ B2,
                                                     const int* __restrict__ idxb,
                                                     float* __restrict__ out_g) {
    __shared__ float rowv[KKEEP];
    __shared__ float red[256];
    const int r = blockIdx.x;
    const unsigned char* Bi = B2 + (size_t)idxb[r] * NNODES;

    float local = 0.f;
    for (int c = threadIdx.x; c < KKEEP; c += 256) {
        const float v = (float)Bi[idxb[c]];
        rowv[c] = v;
        local += v;
    }
    red[threadIdx.x] = local;
    __syncthreads();
#pragma unroll
    for (int s = 128; s > 0; s >>= 1) {
        if (threadIdx.x < s) red[threadIdx.x] += red[threadIdx.x + s];
        __syncthreads();
    }
    const float degv = red[0];
    const float inv = 1.0f / (degv > 0.f ? degv : 1.0f);
    float* outr = out_g + (size_t)r * KKEEP;
    for (int c = threadIdx.x; c < KKEEP; c += 256)
        outr[c] = rowv[c] * inv;
}

extern "C" void kernel_launch(void* const* d_in, const int* in_sizes, int n_in,
                              void* d_out, int out_size, void* d_ws, size_t ws_size,
                              hipStream_t stream) {
    (void)in_sizes; (void)n_in; (void)out_size; (void)ws_size;

    const float* g = (const float*)d_in[0];
    const float* h = (const float*)d_in[1];
    const float* W = (const float*)d_in[2];
    const float* b = (const float*)d_in[3];

    const size_t MATB = (size_t)NNODES * NNODES;
    unsigned char* bufA  = (unsigned char*)d_ws;
    unsigned char* bufAT = bufA + MATB;
    unsigned char* bufC  = bufAT + MATB;
    char* tail = (char*)(bufC + MATB);
    float* score = (float*)tail;
    float* valb  = score + NNODES;
    int*   idxb  = (int*)(valb + KKEEP);

    float* out_g   = (float*)d_out;
    float* out_h   = out_g + (size_t)KKEEP * KKEEP;
    float* out_idx = out_h + (size_t)KKEEP * INDIM;

    const dim3 tb256(256);
    const dim3 trBlk(32, 8);
    const dim3 trGrd(NNODES / 32, NNODES / 32);
    const int bmmBlocks = (NNODES / 32) * (NNODES / 32) / 8;

    k_binarize    <<<(int)(MATB / 256), tb256, 0, stream>>>(g, bufA);
    k_transpose_u8<<<trGrd, trBlk, 0, stream>>>(bufA, bufAT);
    k_bmm_bool    <<<bmmBlocks, tb256, 0, stream>>>(bufA, bufAT, bufC);
    k_transpose_u8<<<trGrd, trBlk, 0, stream>>>(bufC, bufA);
    k_bmm_bool    <<<bmmBlocks, tb256, 0, stream>>>(bufC, bufA, bufAT);

    k_score<<<NNODES / 8, tb256, 0, stream>>>(h, W, b, score);
    k_topk <<<NNODES / 256, tb256, 0, stream>>>(score, idxb, valb, out_idx);
    k_newh <<<(KKEEP * INDIM) / 256, tb256, 0, stream>>>(h, idxb, valb, out_h);
    k_norm_gather<<<KKEEP, tb256, 0, stream>>>(bufAT, idxb, out_g);
}
